// MoE_60361470378718
// MI455X (gfx1250) — matrix operand order repaired and checked by reading, not compiled, not run on hardware
//
#include <hip/hip_runtime.h>
#include <hip/hip_bf16.h>

// ---------------- types ----------------
typedef __attribute__((ext_vector_type(16))) __bf16 v16bf;
typedef __attribute__((ext_vector_type(8)))  float  v8f;

union FragU { uint4 q[2]; v16bf v; };   // 32 bytes: 16 bf16 per lane

// fp32 -> bf16 (round-to-nearest-even)
__device__ __forceinline__ unsigned short f2bf(float f) {
    unsigned u = __float_as_uint(f);
    unsigned r = u + 0x7FFFu + ((u >> 16) & 1u);
    return (unsigned short)(r >> 16);
}

__device__ __forceinline__ float gelu_f(float x) {
    const float c = 0.7978845608028654f;           // sqrt(2/pi)
    float x3 = x * x * x;
    return 0.5f * x * (1.0f + tanhf(c * (x + 0.044715f * x3)));
}

__device__ __forceinline__ void sched_fence() {
#if __has_builtin(__builtin_amdgcn_sched_barrier)
    __builtin_amdgcn_sched_barrier(0);   // keep staging stores ahead of WMMA block
#endif
}

// ---------------- constants ----------------
#define T_TOK   8192
#define D_MODEL 1024
#define D_COND  256
#define D_IN    (D_MODEL + D_COND)   // 1280
#define D_FF    4096
#define N_EXP   8
#define TOPK    2
#define CAP     2560                 // int(1.25 * 2 * 8192 / 8)

// =======================================================================
// 1) Router: logits = x @ Wr, softmax, top-2, normalized weights
//    one wave32 per token
// =======================================================================
__global__ void router_kernel(const float* __restrict__ x,
                              const float* __restrict__ Wr,
                              int* __restrict__ topk, float* __restrict__ wn) {
    int tid = threadIdx.x, lane = tid & 31, wid = tid >> 5;
    int t = blockIdx.x * 8 + wid;
    if (t >= T_TOK) return;
    float acc[N_EXP];
#pragma unroll
    for (int e = 0; e < N_EXP; ++e) acc[e] = 0.f;
    const float* xr = x + (size_t)t * D_MODEL;
    for (int i = lane; i < D_MODEL; i += 32) {
        float xv = xr[i];
        const float* w = Wr + (size_t)i * N_EXP;
#pragma unroll
        for (int e = 0; e < N_EXP; ++e) acc[e] += xv * w[e];
    }
#pragma unroll
    for (int e = 0; e < N_EXP; ++e)
        for (int off = 16; off; off >>= 1) acc[e] += __shfl_xor(acc[e], off, 32);
    if (lane == 0) {
        float mx = acc[0];
#pragma unroll
        for (int e = 1; e < N_EXP; ++e) mx = fmaxf(mx, acc[e]);
        float p[N_EXP];
#pragma unroll
        for (int e = 0; e < N_EXP; ++e) p[e] = __expf(acc[e] - mx);
        int e0 = 0;
#pragma unroll
        for (int e = 1; e < N_EXP; ++e) if (p[e] > p[e0]) e0 = e;   // lowest idx on tie
        int e1 = (e0 == 0) ? 1 : 0;
#pragma unroll
        for (int e = 0; e < N_EXP; ++e)
            if (e != e0 && p[e] > p[e1]) e1 = e;
        float w0 = p[e0], w1 = p[e1];
        float inv = 1.f / (w0 + w1);
        topk[2 * t]     = e0;  topk[2 * t + 1] = e1;
        wn[2 * t]       = w0 * inv;
        wn[2 * t + 1]   = w1 * inv;
    }
}

// =======================================================================
// 2) Binning: deterministic stable-order slot assignment per expert.
//    One block per expert scans items in (t*K+k) order -> matches the
//    reference's stable argsort + cumsum exactly (incl. capacity drops).
// =======================================================================
__global__ void binning_kernel(const int* __restrict__ topk,
                               int* __restrict__ map,        // [E*CAP] slot -> token
                               int* __restrict__ item_gslot, // [T*K]   item -> global slot or -1
                               int* __restrict__ counts) {
    int e = blockIdx.x;
    __shared__ int s_base;
    __shared__ int s_wtot[8];
    int tid = threadIdx.x, lane = tid & 31, wid = tid >> 5;
    if (tid == 0) s_base = 0;
    __syncthreads();
    const int TK = T_TOK * TOPK;
    for (int chunk = 0; chunk < TK; chunk += 256) {
        int item = chunk + tid;
        bool flag = (topk[item] == e);
        unsigned mm = (unsigned)__ballot(flag);
        int prefix = __popc(mm & ((1u << lane) - 1u));
        if (lane == 0) s_wtot[wid] = __popc(mm);
        __syncthreads();
        int woff = 0, tot = 0;
#pragma unroll
        for (int w = 0; w < 8; ++w) {
            if (w < wid) woff += s_wtot[w];
            tot += s_wtot[w];
        }
        if (flag) {
            int slot = s_base + woff + prefix;
            item_gslot[item] = (slot < CAP) ? (e * CAP + slot) : -1;
            if (slot < CAP) map[e * CAP + slot] = item >> 1;   // token = item / K
        }
        __syncthreads();
        if (tid == 0) s_base += tot;
        __syncthreads();
    }
    if (tid == 0) counts[e] = (s_base < CAP) ? s_base : CAP;
}

// =======================================================================
// 3) Build bf16 A matrices: concat(x, cond) for shared path and gathered
//    per-expert rows (zero-filled beyond count -> gelu(0)=0 chain).
// =======================================================================
__global__ void build_shared_A(const float* __restrict__ x,
                               const float* __restrict__ cond,
                               unsigned short* __restrict__ As) {
    int t = blockIdx.x;
    unsigned short* dst = As + (size_t)t * D_IN;
    const float* xr = x + (size_t)t * D_MODEL;
    const float* cr = cond + (size_t)t * D_COND;
    for (int j = threadIdx.x; j < D_IN; j += blockDim.x)
        dst[j] = f2bf(j < D_MODEL ? xr[j] : cr[j - D_MODEL]);
}

__global__ void gather_expert_A(const float* __restrict__ x,
                                const float* __restrict__ cond,
                                const int* __restrict__ map,
                                const int* __restrict__ counts,
                                unsigned short* __restrict__ Ag) {
    int r = blockIdx.x;
    int e = r / CAP, s = r - e * CAP;
    unsigned short* dst = Ag + (size_t)r * D_IN;
    if (s < counts[e]) {
        int t = map[r];
        const float* xr = x + (size_t)t * D_MODEL;
        const float* cr = cond + (size_t)t * D_COND;
        for (int j = threadIdx.x; j < D_IN; j += blockDim.x)
            dst[j] = f2bf(j < D_MODEL ? xr[j] : cr[j - D_MODEL]);
    } else {
        for (int j = threadIdx.x; j < D_IN; j += blockDim.x) dst[j] = 0;
    }
}

// =======================================================================
// 4) Transpose + convert weights fp32[K,N] -> bf16[N,K] (one-time pass so
//    GEMM B-fragments are contiguous ds_load_b128s).  block (32,8).
// =======================================================================
__global__ void transpose_cvt_kernel(const float* __restrict__ in,
                                     unsigned short* __restrict__ out,
                                     int Kd, int N) {
    const float* src = in + (size_t)blockIdx.z * Kd * N;
    unsigned short* dst = out + (size_t)blockIdx.z * Kd * N;
    __shared__ unsigned short tile[32][33];
    int tx = threadIdx.x, ty = threadIdx.y;
    int n0 = blockIdx.x * 32, k0 = blockIdx.y * 32;
#pragma unroll
    for (int i = 0; i < 32; i += 8)
        tile[ty + i][tx] = f2bf(src[(size_t)(k0 + ty + i) * N + n0 + tx]);
    __syncthreads();
#pragma unroll
    for (int i = 0; i < 32; i += 8)
        dst[(size_t)(n0 + ty + i) * Kd + k0 + tx] = tile[tx][ty + i];
}

// =======================================================================
// 5) bf16 WMMA GEMM: C[M,N] = act(A[M,K] @ Bt[N,K]^T)
//    Block tile 256x128x32, 256 threads = 8 waves (4 M x 2 N), each wave
//    64x64 = 4x4 wmma tiles.  Double-buffered LDS, ONE barrier per K-step.
//    Staged tile is stored to LDS *before* the WMMA block (short register
//    lifetimes -> no scratch spills); a sched_barrier pins that order.
//    LDS row stride 40 shorts (80B, 16B aligned, bank-conflict padded).
// =======================================================================
template <bool GELU, bool OUTBF>
__global__ __launch_bounds__(256)
void gemm_bf16_wmma(const unsigned short* __restrict__ A, long long strideAz,
                    const unsigned short* __restrict__ Bt, long long strideBz,
                    void* __restrict__ Cv, long long strideCz,
                    int Kd, int ldC) {
    A  += (size_t)blockIdx.z * strideAz;
    Bt += (size_t)blockIdx.z * strideBz;
    unsigned short* Cb = (unsigned short*)Cv + (size_t)blockIdx.z * strideCz;
    float*          Cf = (float*)Cv          + (size_t)blockIdx.z * strideCz;

    const int blockM = blockIdx.x * 256, blockN = blockIdx.y * 128;
    __shared__ unsigned short lsA[2][256 * 40];   // 2 x 20 KB
    __shared__ unsigned short lsB[2][128 * 40];   // 2 x 10 KB

    int tid = threadIdx.x, lane = tid & 31, wid = tid >> 5;
    int waveM = (wid & 3) * 64, waveN = (wid >> 2) * 64;

    v8f acc[4][4];
    const v8f vzero = {0.f, 0.f, 0.f, 0.f, 0.f, 0.f, 0.f, 0.f};
#pragma unroll
    for (int i = 0; i < 4; ++i)
#pragma unroll
        for (int j = 0; j < 4; ++j) acc[i][j] = vzero;

    // cooperative loads: A = 1 row/thread (4 x b128); B = half row/thread (2 x b128)
    const unsigned short* gA = A + (size_t)(blockM + tid) * Kd;
    int brow = tid >> 1, bkh = (tid & 1) * 16;
    const unsigned short* gB = Bt + (size_t)(blockN + brow) * Kd + bkh;

    // fragment LDS addressing per ISA 7.12.2 bf16 layouts
    int rA = (lane & 15) * 40;
    int kA = (lane < 16) ? 0 : 8;    // A: lanes 0-15 K 0-7 & 16-23; lanes 16-31 K 8-15 & 24-31
    int kB = (lane < 16) ? 0 : 8;    // B: lanes 0-15 K 0-7 & 16-23; lanes 16-31 K 8-15 & 24-31

    // ---- prologue: stage tile 0 into buffer 0 ----
    {
        uint4 ar[4], br[2];
#pragma unroll
        for (int i = 0; i < 4; ++i) ar[i] = *(const uint4*)(gA + i * 8);
        br[0] = *(const uint4*)(gB);
        br[1] = *(const uint4*)(gB + 8);
        unsigned short* dA = &lsA[0][tid * 40];
        unsigned short* dB = &lsB[0][brow * 40 + bkh];
#pragma unroll
        for (int i = 0; i < 4; ++i) *(uint4*)(dA + i * 8) = ar[i];
        *(uint4*)dB = br[0];
        *(uint4*)(dB + 8) = br[1];
    }
    __syncthreads();

    for (int k0 = 0; k0 < Kd; k0 += 32) {
        const int buf = (k0 >> 5) & 1;
        const bool more = (k0 + 32) < Kd;

        // (1) issue this tile's fragment loads first (DS in-order: WMMA's
        //     dscnt wait only needs these, not the later stores)
        FragU af[4], bfr[4];
#pragma unroll
        for (int tm = 0; tm < 4; ++tm) {
            const unsigned short* p = &lsA[buf][(waveM + tm * 16) * 40 + rA + kA];
            af[tm].q[0] = *(const uint4*)p;          // K base
            af[tm].q[1] = *(const uint4*)(p + 16);   // K base + 16
        }
#pragma unroll
        for (int tn = 0; tn < 4; ++tn) {
            const unsigned short* p = &lsB[buf][(waveN + tn * 16) * 40 + rA + kB];
            bfr[tn].q[0] = *(const uint4*)p;         // K base
            bfr[tn].q[1] = *(const uint4*)(p + 16);  // K base + 16
        }

        // (2) next tile: global load -> immediate LDS store into buf^1
        //     (short reg lifetime, safe: buf^1 readers synced last barrier)
        if (more) {
            uint4 ar[4], br[2];
#pragma unroll
            for (int i = 0; i < 4; ++i) ar[i] = *(const uint4*)(gA + k0 + 32 + i * 8);
            br[0] = *(const uint4*)(gB + k0 + 32);
            br[1] = *(const uint4*)(gB + k0 + 40);
            __builtin_prefetch(gA + k0 + 64, 0, 1);   // -> global_prefetch_b8, 2 tiles ahead
            __builtin_prefetch(gB + k0 + 64, 0, 1);
            unsigned short* dA = &lsA[buf ^ 1][tid * 40];
            unsigned short* dB = &lsB[buf ^ 1][brow * 40 + bkh];
#pragma unroll
            for (int i = 0; i < 4; ++i) *(uint4*)(dA + i * 8) = ar[i];
            *(uint4*)dB = br[0];
            *(uint4*)(dB + 8) = br[1];
        }
        sched_fence();   // don't let the WMMA block get hoisted above the stores

        // (3) 16 WMMAs on the current buffer's fragments
#pragma unroll
        for (int tm = 0; tm < 4; ++tm)
#pragma unroll
            for (int tn = 0; tn < 4; ++tn)
                acc[tm][tn] = __builtin_amdgcn_wmma_f32_16x16x32_bf16(
                    false, af[tm].v, false, bfr[tn].v,
                    (short)0, acc[tm][tn], false, false);

        __syncthreads();   // single barrier per K-step (double buffer makes it safe)
    }

    // store: C 16x16 layout — lanes 0-15: M=r, lanes 16-31: M=r+8; N=lane&15
    int rowoff = (lane < 16) ? 0 : 8;
    int coll = lane & 15;
#pragma unroll
    for (int tm = 0; tm < 4; ++tm)
#pragma unroll
        for (int tn = 0; tn < 4; ++tn) {
            int col = blockN + waveN + tn * 16 + coll;
#pragma unroll
            for (int r = 0; r < 8; ++r) {
                int row = blockM + waveM + tm * 16 + rowoff + r;
                float v = acc[tm][tn][r];
                if (GELU) v = gelu_f(v);
                if (OUTBF) Cb[(size_t)row * ldC + col] = f2bf(v);
                else       Cf[(size_t)row * ldC + col] = v;
            }
        }
}

// =======================================================================
// 6) Combine: out = mask * (Ys + K * sum_k w_k * Ye[slot_k]) / (K+1)
// =======================================================================
__global__ void combine_kernel(const float* __restrict__ Ys,
                               const float* __restrict__ Ye,
                               const int* __restrict__ gslot,
                               const float* __restrict__ wn,
                               const float* __restrict__ mask,
                               float* __restrict__ out) {
    int t = blockIdx.x;
    int g0 = gslot[2 * t], g1 = gslot[2 * t + 1];
    float w0 = (g0 >= 0) ? wn[2 * t]     : 0.f;
    float w1 = (g1 >= 0) ? wn[2 * t + 1] : 0.f;
    int s0 = (g0 >= 0) ? g0 : 0;
    int s1 = (g1 >= 0) ? g1 : 0;
    float m = mask[t] * (1.0f / 3.0f);
    const float* y0 = Ye + (size_t)s0 * D_MODEL;
    const float* y1 = Ye + (size_t)s1 * D_MODEL;
    const float* ys = Ys + (size_t)t * D_MODEL;
    float* o = out + (size_t)t * D_MODEL;
    for (int j = threadIdx.x; j < D_MODEL; j += blockDim.x)
        o[j] = m * (ys[j] + 2.0f * (w0 * y0[j] + w1 * y1[j]));
}

// =======================================================================
// launch
// =======================================================================
extern "C" void kernel_launch(void* const* d_in, const int* in_sizes, int n_in,
                              void* d_out, int out_size, void* d_ws, size_t ws_size,
                              hipStream_t stream) {
    const float* x    = (const float*)d_in[0];
    const float* cond = (const float*)d_in[1];
    const float* mask = (const float*)d_in[2];
    const float* Wr   = (const float*)d_in[3];
    const float* W1s  = (const float*)d_in[4];
    const float* W2s  = (const float*)d_in[5];
    const float* W1e  = (const float*)d_in[6];
    const float* W2e  = (const float*)d_in[7];
    float* out = (float*)d_out;

    char* ws = (char*)d_ws;
    size_t off = 0;
    auto carve = [&](size_t bytes) -> void* {
        void* p = ws + off;
        off = (off + bytes + 255) & ~(size_t)255;
        return p;
    };

    int*            topk   = (int*)carve((size_t)T_TOK * TOPK * 4);
    float*          wn     = (float*)carve((size_t)T_TOK * TOPK * 4);
    int*            counts = (int*)carve(N_EXP * 4);
    int*            map    = (int*)carve((size_t)N_EXP * CAP * 4);
    int*            gslot  = (int*)carve((size_t)T_TOK * TOPK * 4);
    unsigned short* As     = (unsigned short*)carve((size_t)T_TOK * D_IN * 2);
    unsigned short* Ag     = (unsigned short*)carve((size_t)N_EXP * CAP * D_IN * 2);
    unsigned short* W1st   = (unsigned short*)carve((size_t)D_IN * D_FF * 2);
    unsigned short* W2st   = (unsigned short*)carve((size_t)D_FF * D_MODEL * 2);
    unsigned short* W1et   = (unsigned short*)carve((size_t)N_EXP * D_IN * D_FF * 2);
    unsigned short* W2et   = (unsigned short*)carve((size_t)N_EXP * D_FF * D_MODEL * 2);
    unsigned short* Hs     = (unsigned short*)carve((size_t)T_TOK * D_FF * 2);
    unsigned short* He     = (unsigned short*)carve((size_t)N_EXP * CAP * D_FF * 2);
    float*          Ys     = (float*)carve((size_t)T_TOK * D_MODEL * 4);
    float*          Ye     = (float*)carve((size_t)N_EXP * CAP * D_MODEL * 4);

    // 1) router + 2) deterministic binning
    router_kernel<<<T_TOK / 8, 256, 0, stream>>>(x, Wr, topk, wn);
    binning_kernel<<<N_EXP, 256, 0, stream>>>(topk, map, gslot, counts);

    // 3) activation staging (fp32 -> bf16 concat)
    build_shared_A<<<T_TOK, 256, 0, stream>>>(x, cond, As);
    gather_expert_A<<<N_EXP * CAP, 256, 0, stream>>>(x, cond, map, counts, Ag);

    // 4) weight transpose+convert (one pass; GEMMs then read bf16 [N,K])
    dim3 tb(32, 8);
    transpose_cvt_kernel<<<dim3(D_FF / 32, D_IN / 32, 1), tb, 0, stream>>>(W1s, W1st, D_IN, D_FF);
    transpose_cvt_kernel<<<dim3(D_MODEL / 32, D_FF / 32, 1), tb, 0, stream>>>(W2s, W2st, D_FF, D_MODEL);
    transpose_cvt_kernel<<<dim3(D_FF / 32, D_IN / 32, N_EXP), tb, 0, stream>>>(W1e, W1et, D_IN, D_FF);
    transpose_cvt_kernel<<<dim3(D_MODEL / 32, D_FF / 32, N_EXP), tb, 0, stream>>>(W2e, W2et, D_FF, D_MODEL);

    // 5) shared-expert MLP (block tile 256x128)
    gemm_bf16_wmma<true, true><<<dim3(T_TOK / 256, D_FF / 128, 1), 256, 0, stream>>>(
        As, 0, W1st, 0, Hs, 0, D_IN, D_FF);
    gemm_bf16_wmma<false, false><<<dim3(T_TOK / 256, D_MODEL / 128, 1), 256, 0, stream>>>(
        Hs, 0, W2st, 0, Ys, 0, D_FF, D_MODEL);

    // 6) grouped expert MLPs (z = expert)
    gemm_bf16_wmma<true, true><<<dim3(CAP / 256, D_FF / 128, N_EXP), 256, 0, stream>>>(
        Ag, (long long)CAP * D_IN, W1et, (long long)D_IN * D_FF,
        He, (long long)CAP * D_FF, D_IN, D_FF);
    gemm_bf16_wmma<false, false><<<dim3(CAP / 256, D_MODEL / 128, N_EXP), 256, 0, stream>>>(
        He, (long long)CAP * D_FF, W2et, (long long)D_FF * D_MODEL,
        Ye, (long long)CAP * D_MODEL, D_FF, D_MODEL);

    // 7) weighted scatter-combine
    combine_kernel<<<T_TOK, 256, 0, stream>>>(Ys, Ye, gslot, wn, mask, out);
}
